// REM_87213605913032
// MI455X (gfx1250) — matrix operand order repaired and checked by reading, not compiled, not run on hardware
//
#include <hip/hip_runtime.h>
#include <hip/hip_bf16.h>

// ---------------------------------------------------------------------------
// CDNA5 (gfx1250) implementation of the equivariant-GNN reference.
// wave32, WMMA bf16 16x16x32 for all large GEMM-like contractions,
// fragment-major ("swizzled") operand layouts so every WMMA operand is a
// single aligned 32B-per-lane load (2x b128), fp32 global atomics for the
// edge->node scatter, fp32 VALU for tiny irregular einsums.
// ---------------------------------------------------------------------------

typedef __bf16 bf16;
typedef __attribute__((ext_vector_type(16))) __bf16 v16bf;
typedef __attribute__((ext_vector_type(8)))  float  v8f;

#define NN   8192      // nodes
#define BBG  32        // graphs
#define EE   32768     // edges

__device__ __forceinline__ bf16 f2bf(float f) { return (bf16)f; }
__device__ __forceinline__ v8f  vzero8() {
  v8f z = {0.f,0.f,0.f,0.f,0.f,0.f,0.f,0.f};
  return z;
}

// ---- fragment-major layouts -----------------------------------------------
// A-matrix 16x32 bf16 fragment (ISA 7.12.2): lane = hf*16 + m, element j maps
// to klo = ((j&8)<<1) | (hf<<3) | (j&7).  Storage: [kt][lane][j] contiguous.
__device__ __forceinline__ int a_klo(int hf, int j) {
  return ((j & 8) << 1) | (hf << 3) | (j & 7);
}
// inverse (for writing D results as next A operand): given klo,
// hf = (klo>>3)&1 ; j = ((klo&16)>>1)|(klo&7)
__device__ __forceinline__ int a_j_of_klo(int klo) {
  return ((klo & 16) >> 1) | (klo & 7);
}

// B-matrix 32x16 fragment: lane = hf*16 + n_local, element j maps to
// k = kt*32 + a_klo(hf, j) (same as A).  Storage: [kt][nt][lane][j] contiguous.

__device__ __forceinline__ v8f wmma_bf16(v16bf a, v16bf b, v8f c) {
  return __builtin_amdgcn_wmma_f32_16x16x32_bf16(false, a, false, b, (short)0, c, false, false);
}

__device__ __forceinline__ v16bf ldfrag(const bf16* p) {
  return *(const v16bf*)p;   // 32B aligned -> 2x b128
}

__device__ __forceinline__ void real_sh_dev(float x, float y, float z, float* Y) {
  const float s3 = 1.7320508075688772f, s5 = 2.23606797749979f, s15 = 3.872983346207417f;
  Y[0] = 1.f;          Y[1] = s3 * y;      Y[2] = s3 * z;      Y[3] = s3 * x;
  Y[4] = s15 * x * y;  Y[5] = s15 * y * z;
  Y[6] = 0.5f * s5 * (3.f * z * z - 1.f);
  Y[7] = s15 * x * z;  Y[8] = 0.5f * s15 * (x * x - y * y);
  Y[9]  = 2.0916500663351889f * y * (3.f * x * x - y * y);   // sqrt(35/8)
  Y[10] = 10.246950765959598f * x * y * z;                    // sqrt(105)
  Y[11] = 1.6201851746019651f * y * (5.f * z * z - 1.f);      // sqrt(21/8)
  Y[12] = 0.5f * 2.6457513110645907f * (5.f * z * z * z - 3.f * z);
  Y[13] = 1.6201851746019651f * x * (5.f * z * z - 1.f);
  Y[14] = 0.5f * 10.246950765959598f * (x * x - y * y) * z;
  Y[15] = 2.0916500663351889f * x * (x * x - 3.f * y * y);
}

// ------------------------------- kernels -----------------------------------

__global__ void k_zero(float* p, int n) {
  for (int i = blockIdx.x * blockDim.x + threadIdx.x; i < n; i += gridDim.x * blockDim.x)
    p[i] = 0.f;
}

// Repack weights into bf16 fragment-major layouts.
// B1p/B2p : K=1024 (index (khid,f)) x N=128, tiles [kt:32][nt:8]
// Bself   : K=64 x N=128, tiles [kt:2][nt:8]
// Gt      : K=96 (zero pad >=84) x N=4096, tiles [kt:3][nt:256]
// Gf      : K=4096 x N=96 (zero pad >=84), tiles [kt:128][nt:6]
__global__ void k_prep_weights(const float* W2a, const float* W2b,
                               const float* Gto, const float* Gfrom,
                               bf16* B1p, bf16* B2p, bf16* Bself, bf16* Gt, bf16* Gf) {
  const int s0 = 1024 * 128;               // B1p
  const int s1 = s0 + 1024 * 128;          // B2p
  const int s2 = s1 + 64 * 128;            // Bself
  const int s3 = s2 + 96 * 4096;           // Gt
  const int s4 = s3 + 4096 * 96;           // Gf
  for (int i = blockIdx.x * blockDim.x + threadIdx.x; i < s4; i += gridDim.x * blockDim.x) {
    if (i < s2) {  // the three radial-weight matrices share tile decode (nt=8)
      int idx, which;
      if (i < s0)      { idx = i;      which = 0; }
      else if (i < s1) { idx = i - s0; which = 1; }
      else             { idx = i - s1; which = 2; }
      int tile = idx >> 9, rem = idx & 511;
      int lane = rem >> 4, j = rem & 15;
      int kt = tile >> 3, nt = tile & 7;
      int k = kt * 32 + a_klo(lane >> 4, j);
      int n = nt * 16 + (lane & 15);
      float v;
      if (which == 0) { int kh = k >> 4, f = k & 15; v = W2a[kh * 2048 + f * 128 + n]; }
      else if (which == 1) { int kh = k >> 4, f = k & 15; v = W2b[kh * 2176 + f * 128 + n]; }
      else { v = W2b[k * 2176 + 2048 + n]; }
      (which == 0 ? B1p : which == 1 ? B2p : Bself)[idx] = f2bf(v);
    } else if (i < s3) {  // Gt
      int idx = i - s2;
      int tile = idx >> 9, rem = idx & 511;
      int lane = rem >> 4, j = rem & 15;
      int kt = tile >> 8, nt = tile & 255;
      int k = kt * 32 + a_klo(lane >> 4, j);
      int n = nt * 16 + (lane & 15);
      Gt[idx] = f2bf(k < 84 ? Gto[k * 4096 + n] : 0.f);
    } else {              // Gf
      int idx = i - s3;
      int tile = idx >> 9, rem = idx & 511;
      int lane = rem >> 4, j = rem & 15;
      int kt = tile / 6, nt = tile % 6;
      int k = kt * 32 + a_klo(lane >> 4, j);
      int n = nt * 16 + (lane & 15);
      Gf[idx] = f2bf(n < 84 ? Gfrom[(size_t)k * 84 + n] : 0.f);
    }
  }
}

// Per edge: geometry, SH, cutoff-scaled radial hidden activations (bf16), src feats.
__global__ void k_edge_prep(const float* nf, const float* pos,
                            const float* r1W1, const float* r1b,
                            const float* r2W1, const float* r2b,
                            const int* esrc, const int* edst,
                            bf16* a1, bf16* a2, float* Yb, bf16* nfs) {
  int e = blockIdx.x * blockDim.x + threadIdx.x;
  if (e >= EE) return;
  int s = esrc[e], t = edst[e];
  float ex = pos[t * 3 + 0] - pos[s * 3 + 0];
  float ey = pos[t * 3 + 1] - pos[s * 3 + 1];
  float ez = pos[t * 3 + 2] - pos[s * 3 + 2];
  float d = sqrtf(ex * ex + ey * ey + ez * ez) + 1e-9f;
  float x = ex / d, y = ey / d, z = ez / d;
  float Y[16];
  real_sh_dev(x, y, z, Y);
#pragma unroll
  for (int c = 0; c < 16; ++c) Yb[(size_t)e * 16 + c] = Y[c];
  float u = d / 3.5f;
  float cut = (u < 1.f) ? 0.5f * (__cosf(3.14159265358979f * u) + 1.f) : 0.f;
#pragma unroll 4
  for (int k = 0; k < 64; ++k) {
    float h1 = fmaxf(u * r1W1[k] + r1b[k], 0.f);
    float h2 = fmaxf(u * r2W1[k] + r2b[k], 0.f);
    a1[(size_t)e * 64 + k] = f2bf(h1 * cut);
    a2[(size_t)e * 64 + k] = f2bf(h2 * cut);
  }
#pragma unroll
  for (int f = 0; f < 16; ++f) nfs[(size_t)e * 16 + f] = f2bf(nf[(size_t)s * 16 + f]);
}

// Layer-1 message GEMM + scatter. A' = outer(a1, nf_src) per edge (K=1024),
// built straight into fragment-major LDS; t1 = A' @ B1p (128 cols);
// h1[dst] += t1[.,LIDX]*Y / (sqrt(F)*deg)
__global__ void __launch_bounds__(128) k_msg1(const bf16* a1, const bf16* nfs, const float* Yb,
                                              const int* edst, const bf16* B1p, float* h1) {
  __shared__ bf16  Ash[16 * 1024];   // fragment-major: [kt:32][lane:32][j:16]
  __shared__ bf16  a1sh[16 * 64];
  __shared__ bf16  nfsh[16 * 16];
  __shared__ float Ysh[16 * 16];
  __shared__ int   dsh[16];
  int tid = threadIdx.x;
  int e0  = blockIdx.x * 16;
  for (int i = tid; i < 16 * 64; i += 128) a1sh[i] = a1[(size_t)e0 * 64 + i];
  for (int i = tid; i < 16 * 16; i += 128) { nfsh[i] = nfs[(size_t)e0 * 16 + i]; Ysh[i] = Yb[(size_t)e0 * 16 + i]; }
  if (tid < 16) dsh[tid] = edst[e0 + tid];
  __syncthreads();
  for (int i = tid; i < 16 * 1024; i += 128) {
    int kt = i >> 9, rem = i & 511, lane2 = rem >> 4, j = rem & 15;
    int m = lane2 & 15, hf = lane2 >> 4;
    int K = kt * 32 + a_klo(hf, j);          // K-index (khid,f)
    int kh = K >> 4, f = K & 15;
    Ash[i] = f2bf((float)a1sh[m * 64 + kh] * (float)nfsh[m * 16 + f]);
  }
  __syncthreads();
  int lane = tid & 31, wid = tid >> 5;
  const int CSTART[4] = {0, 1, 4, 9};
  for (int nt = wid; nt < 8; nt += 4) {
    v8f acc = vzero8();
    for (int kt = 0; kt < 32; ++kt) {
      v16bf af = ldfrag(Ash + ((kt * 32 + lane) << 4));
      v16bf bf = ldfrag(B1p + (((size_t)(kt * 8 + nt) * 32 + lane) << 4));
      acc = wmma_bf16(af, bf, acc);
    }
    int n = nt * 16 + (lane & 15);
    int g = n >> 3, p = (n >> 2) & 1, l = n & 3;
    int hfp = lane >> 4;
    int nc = 2 * l + 1, c0 = CSTART[l];
#pragma unroll
    for (int r = 0; r < 8; ++r) {
      int m = r + 8 * hfp;
      float val = acc[r] * 0.125f;              // /sqrt(16) /deg(2)
      size_t base = (size_t)dsh[m] * 512 + g * 32 + p * 16;
      for (int t = 0; t < nc; ++t)
        atomicAdd(&h1[base + c0 + t], val * Ysh[m * 16 + c0 + t]);
    }
  }
}

// In-place gate: rows = (n,f,p), 16 cols each.
__global__ void k_gate(float* h, int rows) {
  int r = blockIdx.x * blockDim.x + threadIdx.x;
  if (r >= rows) return;
  float* row = h + (size_t)r * 16;
  float s  = row[0];
  float sg = 1.f / (1.f + __expf(-s));
  row[0] = fmaxf(s, 0.f);
#pragma unroll
  for (int c = 1; c < 16; ++c) row[c] *= sg;
}

// Layer-2: main GEMM A' = outer(a2, h1g[src][:,0,0]) (K=1024) -> t2,
// self GEMM tself = a2 @ Bself (K=64), scatter both into h2.
__global__ void __launch_bounds__(128) k_msg2(const bf16* a2, const float* h1g, const float* Yb,
                                              const int* esrc, const int* edst,
                                              const bf16* B2p, const bf16* Bself, float* h2) {
  __shared__ bf16  Ash[16 * 1024];   // fragment-major main A'
  __shared__ bf16  a2sh[16 * 64];    // row-major a2 tile
  __shared__ bf16  a2sw[16 * 64];    // fragment-major a2 tile (self GEMM A)
  __shared__ bf16  s1sh[16 * 16];
  __shared__ float Ysh[16 * 16];
  __shared__ int   dsh[16], ssh[16];
  int tid = threadIdx.x;
  int e0  = blockIdx.x * 16;
  if (tid < 16) { dsh[tid] = edst[e0 + tid]; ssh[tid] = esrc[e0 + tid]; }
  for (int i = tid; i < 16 * 64; i += 128) a2sh[i] = a2[(size_t)e0 * 64 + i];
  for (int i = tid; i < 16 * 16; i += 128) Ysh[i] = Yb[(size_t)e0 * 16 + i];
  __syncthreads();
  for (int i = tid; i < 16 * 16; i += 128) {
    int m = i >> 4, f = i & 15;
    s1sh[i] = f2bf(h1g[(size_t)ssh[m] * 512 + f * 32]);   // gated scalar channel
  }
  __syncthreads();
  for (int i = tid; i < 16 * 1024; i += 128) {
    int kt = i >> 9, rem = i & 511, lane2 = rem >> 4, j = rem & 15;
    int m = lane2 & 15, hf = lane2 >> 4;
    int K = kt * 32 + a_klo(hf, j);
    int kh = K >> 4, f = K & 15;
    Ash[i] = f2bf((float)a2sh[m * 64 + kh] * (float)s1sh[m * 16 + f]);
  }
  for (int i = tid; i < 16 * 64; i += 128) {
    int kt = i >> 9, rem = i & 511, lane2 = rem >> 4, j = rem & 15;
    int m = lane2 & 15, hf = lane2 >> 4;
    int K = kt * 32 + a_klo(hf, j);          // < 64
    a2sw[i] = a2sh[m * 64 + K];
  }
  __syncthreads();
  int lane = tid & 31, wid = tid >> 5;
  const int CSTART[4] = {0, 1, 4, 9};
  // main term
  for (int nt = wid; nt < 8; nt += 4) {
    v8f acc = vzero8();
    for (int kt = 0; kt < 32; ++kt) {
      v16bf af = ldfrag(Ash + ((kt * 32 + lane) << 4));
      v16bf bf = ldfrag(B2p + (((size_t)(kt * 8 + nt) * 32 + lane) << 4));
      acc = wmma_bf16(af, bf, acc);
    }
    int n = nt * 16 + (lane & 15);
    int g = n >> 3, p = (n >> 2) & 1, l = n & 3;
    int hfp = lane >> 4;
    int nc = 2 * l + 1, c0 = CSTART[l];
#pragma unroll
    for (int r = 0; r < 8; ++r) {
      int m = r + 8 * hfp;
      float val = acc[r] * 0.125f;
      size_t base = (size_t)dsh[m] * 512 + g * 32 + p * 16;
      for (int t = 0; t < nc; ++t)
        atomicAdd(&h2[base + c0 + t], val * Ysh[m * 16 + c0 + t]);
    }
  }
  // self term
  for (int nt = wid; nt < 8; nt += 4) {
    v8f acc = vzero8();
    for (int kt = 0; kt < 2; ++kt) {
      v16bf af = ldfrag(a2sw + ((kt * 32 + lane) << 4));
      v16bf bf = ldfrag(Bself + (((size_t)(kt * 8 + nt) * 32 + lane) << 4));
      acc = wmma_bf16(af, bf, acc);
    }
    int n = nt * 16 + (lane & 15);
    int f = n >> 3, p = (n >> 2) & 1, l = n & 3;
    int hfp = lane >> 4;
    int nc = 2 * l + 1, c0 = CSTART[l];
#pragma unroll
    for (int r = 0; r < 8; ++r) {
      int m = r + 8 * hfp;
      float val = acc[r] * 0.5f;                 // /deg only
      size_t sb   = (size_t)ssh[m] * 512 + f * 32 + p * 16;
      size_t base = (size_t)dsh[m] * 512 + f * 32 + p * 16;
      for (int t = 0; t < nc; ++t)
        atomicAdd(&h2[base + c0 + t], val * h1g[sb + c0 + t]);
    }
  }
}

// Gate h2 inline + mean-pool each graph's 256 contiguous nodes -> enc (B,512)
__global__ void k_gate_pool(const float* h2, float* enc) {
  int b = blockIdx.x, tid = threadIdx.x;  // 256 threads
  for (int j = tid; j < 512; j += 256) {
    int c = j & 15, sj = j & ~15;
    float acc = 0.f;
    for (int node = 0; node < 256; ++node) {
      size_t base = (size_t)(b * 256 + node) * 512;
      float s = h2[base + sj];
      float v = (c == 0) ? fmaxf(s, 0.f)
                         : h2[base + j] * (1.f / (1.f + __expf(-s)));
      acc += v;
    }
    enc[(size_t)b * 512 + j] = acc * (1.f / 256.f);
  }
}

// z[b,g,d] from enc and Wlin (tiny einsum, VALU)
__global__ void k_zlin(const float* enc, const float* Wlin, float* z) {
  int b = blockIdx.x, tid = threadIdx.x;  // 128 threads
  const int OFF[5]  = {0, 1, 10, 35, 84};
  const int MOFF[4] = {0, 1, 4, 9};
  for (int idx = tid; idx < 16 * 84; idx += 128) {
    int g = idx / 84, d = idx % 84;
    int l = (d < 1) ? 0 : (d < 10) ? 1 : (d < 35) ? 2 : 3;
    int m = 2 * l + 1, loc = d - OFF[l], j = loc / m, c = loc % m;
    int c0 = MOFF[l];
    float acc = 0.f;
    for (int mm = 0; mm < 32; ++mm)
      acc += enc[(size_t)b * 512 + mm * 16 + c0 + c] * Wlin[mm * 256 + g * 16 + c0 + j];
    z[(size_t)b * (16 * 84) + idx] = acc * 0.17677669529663687f;  // 1/sqrt(32)
  }
}

// Generic small SO(3) conv, fp32 VALU (irregular m = 1,3,5,7 blocks)
__global__ void k_conv(const float* X, const float* psi, float* Out, int fi, int fo) {
  int bg = blockIdx.x;
  int b = bg / fo, g = bg % fo;
  int d = threadIdx.x;
  if (d >= 84) return;
  const int OFF[5] = {0, 1, 10, 35, 84};
  int l = (d < 1) ? 0 : (d < 10) ? 1 : (d < 35) ? 2 : 3;
  int m = 2 * l + 1, o0 = OFF[l], loc = d - o0, j = loc / m, c = loc % m;
  float acc = 0.f;
  for (int f = 0; f < fi; ++f) {
    const float* xr = X + ((size_t)b * fi + f) * 84 + o0;
    const float* pr = psi + ((size_t)f * fo + g) * 84 + o0;
    for (int i = 0; i < m; ++i) acc += xr[i * m + c] * pr[i * m + j];
  }
  Out[((size_t)b * fo + g) * 84 + d] = acc * rsqrtf((float)(fi * m));
}

// so3_act, fused: out = relu(X@Gt/sqrt(84)) @ Gf / 64, rows tiled by 16,
// g staged through LDS (fragment-major) in 1024-col chunks.
__global__ void __launch_bounds__(128) k_act(const float* X, float* Out,
                                             const bf16* Gt, const bf16* Gf) {
  __shared__ bf16 Ash[3 * 512];       // fragment-major X tile (K=96)
  __shared__ bf16 Gsh[32 * 512];      // fragment-major g chunk (K=1024)
  int tid = threadIdx.x, lane = tid & 31, wid = tid >> 5;
  int r0 = blockIdx.x * 16;
  for (int i = tid; i < 3 * 512; i += 128) {
    int kt = i >> 9, rem = i & 511, lane2 = rem >> 4, j = rem & 15;
    int m = lane2 & 15, hf = lane2 >> 4;
    int k = kt * 32 + a_klo(hf, j);
    Ash[i] = f2bf(k < 84 ? X[(size_t)(r0 + m) * 84 + k] : 0.f);
  }
  v8f acc2[2];
  acc2[0] = vzero8();
  acc2[1] = vzero8();
  const float sA = rsqrtf(84.f);
  __builtin_prefetch(Gf, 0, 1);
  for (int chunk = 0; chunk < 4; ++chunk) {
    __syncthreads();  // Ash ready (first iter) / Gsh consumed (later iters)
    for (int nt = wid; nt < 64; nt += 4) {
      v8f acc = vzero8();
      for (int kt = 0; kt < 3; ++kt) {
        v16bf af = ldfrag(Ash + ((kt * 32 + lane) << 4));
        v16bf bf = ldfrag(Gt + (((size_t)(kt * 256 + chunk * 64 + nt) * 32 + lane) << 4));
        acc = wmma_bf16(af, bf, acc);
      }
      // write relu(g) as the next GEMM's A-fragments
      int n   = nt * 16 + (lane & 15);   // K-index of second GEMM (within chunk)
      int kt2 = n >> 5, klo = n & 31;
      int hf2 = (klo >> 3) & 1;
      int j2  = a_j_of_klo(klo);
      int hfp = lane >> 4;
#pragma unroll
      for (int r = 0; r < 8; ++r) {
        int m = r + 8 * hfp;
        Gsh[((kt2 * 32 + ((hf2 << 4) | m)) << 4) + j2] = f2bf(fmaxf(acc[r] * sA, 0.f));
      }
    }
    __syncthreads();  // Gsh ready
#pragma unroll
    for (int t = 0; t < 2; ++t) {
      int nt = wid + t * 4;
      if (nt < 6) {
        v8f acc = acc2[t];
        for (int kt = 0; kt < 32; ++kt) {
          v16bf af = ldfrag(Gsh + ((kt * 32 + lane) << 4));
          v16bf bf = ldfrag(Gf + (((size_t)((chunk * 32 + kt) * 6 + nt) * 32 + lane) << 4));
          acc = wmma_bf16(af, bf, acc);
        }
        acc2[t] = acc;
      }
    }
  }
#pragma unroll
  for (int t = 0; t < 2; ++t) {
    int nt = wid + t * 4;
    if (nt < 6) {
      int n = nt * 16 + (lane & 15);
      int hfp = lane >> 4;
      if (n < 84) {
#pragma unroll
        for (int r = 0; r < 8; ++r) {
          int m = r + 8 * hfp;
          Out[(size_t)(r0 + m) * 84 + n] = acc2[t][r] * (1.f / 64.f);  // 1/sqrt(4096)
        }
      }
    }
  }
}

// Final S2 projection + orientation SH dot -> out[b]
__global__ void k_final(const float* x3, const float* Ws2, const float* orient, float* out) {
  __shared__ float red[16];
  int b = blockIdx.x, tid = threadIdx.x;  // 32 threads
  const int OFF[5]  = {0, 1, 10, 35, 84};
  const int MOFF[4] = {0, 1, 4, 9};
  if (tid < 16) {
    int ct = tid;
    int l = (ct < 1) ? 0 : (ct < 4) ? 1 : (ct < 9) ? 2 : 3;
    int m = 2 * l + 1, c = ct - MOFF[l], o0 = OFF[l];
    float acc = 0.f;
    for (int f = 0; f < 256; ++f) {
      const float* xr = x3 + ((size_t)b * 256 + f) * 84 + o0;
      const float* wr = Ws2 + f * 16 + MOFF[l];
      for (int i = 0; i < m; ++i) acc += xr[i * m + c] * wr[i];
    }
    float s2 = acc * rsqrtf(256.f * (float)m);
    float a = orient[b * 2 + 0], rr = orient[b * 2 + 1];
    float sa = __sinf(a), ca = __cosf(a), sr = __sinf(rr), cr = __cosf(rr);
    float Y[16];
    real_sh_dev(-sa * cr, -sa * sr, -ca, Y);
    red[tid] = s2 * Y[ct];
  }
  __syncthreads();
  if (tid == 0) {
    float s = 0.f;
    for (int i = 0; i < 16; ++i) s += red[i];
    out[b] = s;
  }
}

// ------------------------------- launcher ----------------------------------

extern "C" void kernel_launch(void* const* d_in, const int* in_sizes, int n_in,
                              void* d_out, int out_size, void* d_ws, size_t ws_size,
                              hipStream_t stream) {
  (void)in_sizes; (void)n_in; (void)out_size; (void)ws_size;
  const float* node_feat = (const float*)d_in[0];
  const float* pos       = (const float*)d_in[1];
  const float* orient    = (const float*)d_in[2];
  const float* r1W1      = (const float*)d_in[3];
  const float* r1b       = (const float*)d_in[4];
  const float* r1W2      = (const float*)d_in[5];
  const float* r2W1      = (const float*)d_in[6];
  const float* r2b       = (const float*)d_in[7];
  const float* r2W2      = (const float*)d_in[8];
  const float* Wlin      = (const float*)d_in[9];
  const float* psi1      = (const float*)d_in[10];
  const float* psi2      = (const float*)d_in[11];
  const float* psi3      = (const float*)d_in[12];
  const float* Gto       = (const float*)d_in[13];
  const float* Gfrom     = (const float*)d_in[14];
  const float* Ws2       = (const float*)d_in[15];
  const int*   esrc      = (const int*)d_in[16];
  const int*   edst      = (const int*)d_in[17];
  // d_in[18] (batch) implied by contiguous node layout.

  char* w = (char*)d_ws;
  size_t off = 0;
  auto alloc = [&](size_t bytes) -> char* {
    char* p = w + off;
    off = (off + bytes + 255) & ~(size_t)255;
    return p;
  };
  float* h1  = (float*)alloc((size_t)NN * 512 * 4);
  float* h2  = (float*)alloc((size_t)NN * 512 * 4);   // contiguous after h1
  bf16*  a1  = (bf16*) alloc((size_t)EE * 64 * 2);
  bf16*  a2  = (bf16*) alloc((size_t)EE * 64 * 2);
  float* Yb  = (float*)alloc((size_t)EE * 16 * 4);
  bf16*  nfs = (bf16*) alloc((size_t)EE * 16 * 2);
  bf16*  B1p = (bf16*) alloc((size_t)1024 * 128 * 2);
  bf16*  B2p = (bf16*) alloc((size_t)1024 * 128 * 2);
  bf16*  Bsf = (bf16*) alloc((size_t)64 * 128 * 2);
  bf16*  Gt  = (bf16*) alloc((size_t)96 * 4096 * 2);
  bf16*  Gf  = (bf16*) alloc((size_t)4096 * 96 * 2);
  float* enc = (float*)alloc((size_t)BBG * 512 * 4);
  float* z   = (float*)alloc((size_t)BBG * 16 * 84 * 4);
  float* c1  = (float*)alloc((size_t)BBG * 64 * 84 * 4);
  float* x1  = (float*)alloc((size_t)BBG * 64 * 84 * 4);
  float* c2  = (float*)alloc((size_t)BBG * 128 * 84 * 4);
  float* x2  = (float*)alloc((size_t)BBG * 128 * 84 * 4);
  float* c3  = (float*)alloc((size_t)BBG * 256 * 84 * 4);
  float* x3  = (float*)alloc((size_t)BBG * 256 * 84 * 4);

  k_zero<<<2048, 256, 0, stream>>>(h1, 2 * NN * 512);
  k_prep_weights<<<1024, 256, 0, stream>>>(r1W2, r2W2, Gto, Gfrom, B1p, B2p, Bsf, Gt, Gf);
  k_edge_prep<<<EE / 256, 256, 0, stream>>>(node_feat, pos, r1W1, r1b, r2W1, r2b,
                                            esrc, edst, a1, a2, Yb, nfs);
  k_msg1<<<EE / 16, 128, 0, stream>>>(a1, nfs, Yb, edst, B1p, h1);
  k_gate<<<(NN * 32) / 256, 256, 0, stream>>>(h1, NN * 32);
  k_msg2<<<EE / 16, 128, 0, stream>>>(a2, h1, Yb, esrc, edst, B2p, Bsf, h2);
  k_gate_pool<<<BBG, 256, 0, stream>>>(h2, enc);
  k_zlin<<<BBG, 128, 0, stream>>>(enc, Wlin, z);
  k_conv<<<BBG * 64, 96, 0, stream>>>(z, psi1, c1, 16, 64);
  k_act<<<(BBG * 64) / 16, 128, 0, stream>>>(c1, x1, Gt, Gf);
  k_conv<<<BBG * 128, 96, 0, stream>>>(x1, psi2, c2, 64, 128);
  k_act<<<(BBG * 128) / 16, 128, 0, stream>>>(c2, x2, Gt, Gf);
  k_conv<<<BBG * 256, 96, 0, stream>>>(x2, psi3, c3, 128, 256);
  k_act<<<(BBG * 256) / 16, 128, 0, stream>>>(c3, x3, Gt, Gf);
  k_final<<<BBG, 32, 0, stream>>>(x3, Ws2, orient, (float*)d_out);
}
